// DecoderLayer_12850542149815
// MI455X (gfx1250) — matrix operand order repaired and checked by reading, not compiled, not run on hardware
//
#include <hip/hip_runtime.h>
#include <math.h>

// ---------------- problem constants ----------------
#define Bc   2
#define Sc   512
#define Hc   1024
#define HQc  16
#define HKVc 4
#define Dc   64
#define Ic   4096
#define Ec   8
#define Tc   (Bc*Sc)          // 1024 tokens
#define TOPKc 2
#define MAXROWS 2560          // T*TOPK + E*64 padding
#define MAXTILES 40           // MAXROWS/64
#define EPSc 1.1920929e-07f

typedef __bf16 bf16;
typedef __attribute__((ext_vector_type(16))) __bf16 bf16x16;
typedef __attribute__((ext_vector_type(8)))  __bf16 bf16x8;
typedef __attribute__((ext_vector_type(2)))  __bf16 bf16x2;
typedef __attribute__((ext_vector_type(8)))  float  f32x8;

// async global->LDS copy (gfx1250, ASYNCcnt-tracked)
#define ASYNC_CP_B128(ldsoff, gptr) \
  asm volatile("global_load_async_to_lds_b128 %0, %1, off" :: "v"(ldsoff), "v"(gptr) : "memory")
#define WAIT_ASYNC0() asm volatile("s_wait_asynccnt 0x0" ::: "memory")

// ---------------- WMMA helpers ----------------
__device__ __forceinline__ f32x8 wmma_bf16(bf16x16 a, bf16x16 b, f32x8 c) {
  return __builtin_amdgcn_wmma_f32_16x16x32_bf16(false, a, false, b, (short)0, c, false, false);
}

// A fragment (16x32, MxK): lane l holds row M = m_base + (l&15),
// K elements kl..kl+15 contiguous with kl = (l<16)?0:16 (matches load_b_frag).
__device__ __forceinline__ bf16x16 load_a_frag(const bf16* lds, int stride,
                                               int m_base, int k0, int lane) {
  int m  = m_base + (lane & 15);
  int kl = (lane & 16) ? 16 : 0;
  const bf16* base = lds + m * stride + k0 + kl;
  bf16x8 lo = *(const bf16x8*)(base);
  bf16x8 hi = *(const bf16x8*)(base + 8);
  bf16x16 r;
#pragma unroll
  for (int i = 0; i < 8; ++i) { r[i] = lo[i]; r[i + 8] = hi[i]; }
  return r;
}

// B fragment (32x16, KxN) from LDS laid out [n][k]:
// lane l holds col N = n_base + (l&15), K = k0 + (l<16 ? 0..15 : 16..31) contiguous.
__device__ __forceinline__ bf16x16 load_b_frag(const bf16* lds, int stride,
                                               int n_base, int k0, int lane) {
  int n  = n_base + (lane & 15);
  int kh = (lane & 16) ? 16 : 0;
  return *(const bf16x16*)(lds + n * stride + k0 + kh);
}

__device__ __forceinline__ bf16x8 pack8(float4 a, float4 b) {
  bf16x8 v;
  v[0] = (bf16)a.x; v[1] = (bf16)a.y; v[2] = (bf16)a.z; v[3] = (bf16)a.w;
  v[4] = (bf16)b.x; v[5] = (bf16)b.y; v[6] = (bf16)b.z; v[7] = (bf16)b.w;
  return v;
}

// reductions within 16-lane halves (wave32)
__device__ __forceinline__ float half16_max(float v) {
#pragma unroll
  for (int m = 8; m >= 1; m >>= 1) v = fmaxf(v, __shfl_xor(v, m, 32));
  return v;
}
__device__ __forceinline__ float half16_sum(float v) {
#pragma unroll
  for (int m = 8; m >= 1; m >>= 1) v += __shfl_xor(v, m, 32);
  return v;
}

// ---------------- RMSNorm ----------------
__global__ __launch_bounds__(256) void rmsnorm_k(const float* __restrict__ x,
                                                 const float* __restrict__ w,
                                                 float* __restrict__ y) {
  int t = blockIdx.x;
  const float* xr = x + (size_t)t * Hc;
  float ss = 0.f;
  for (int i = threadIdx.x * 4; i < Hc; i += 1024) {
    float4 v = *(const float4*)(xr + i);
    ss += v.x * v.x + v.y * v.y + v.z * v.z + v.w * v.w;
  }
#pragma unroll
  for (int m = 16; m >= 1; m >>= 1) ss += __shfl_xor(ss, m, 32);
  __shared__ float red[8];
  if ((threadIdx.x & 31) == 0) red[threadIdx.x >> 5] = ss;
  __syncthreads();
  float tot = 0.f;
#pragma unroll
  for (int i = 0; i < 8; ++i) tot += red[i];
  float inv = rsqrtf(tot / (float)Hc + EPSc);
  float* yr = y + (size_t)t * Hc;
  for (int i = threadIdx.x * 4; i < Hc; i += 1024) {
    float4 v = *(const float4*)(xr + i);
    float4 g = *(const float4*)(w + i);
    float4 o; o.x = v.x * inv * g.x; o.y = v.y * inv * g.y;
    o.z = v.z * inv * g.z; o.w = v.w * inv * g.w;
    *(float4*)(yr + i) = o;
  }
}

// ---------------- generic dense GEMM: C = A(MxK)@B(KxN) + bias (+resid) ----------------
// block 256 = 8 waves (4 row x 2 col), 64x64 tile, K step 32, bf16 WMMA.
__global__ __launch_bounds__(256) void gemm_bf16_k(const float* __restrict__ A, int lda,
                                                   const float* __restrict__ Bw, int ldb,
                                                   const float* __restrict__ bias,
                                                   const float* __restrict__ resid,
                                                   float* __restrict__ C0,
                                                   float* __restrict__ C1,
                                                   int N, int K) {
  __shared__ __align__(32) bf16 As[64 * 32];   // [m][k]
  __shared__ __align__(32) bf16 Bt[64 * 32];   // [n][k]
  int tid = threadIdx.x, lane = tid & 31, wid = tid >> 5;
  int wrow = wid >> 1, wcol = wid & 1;
  int m0 = blockIdx.y * 64, n0 = blockIdx.x * 64;

  // staging maps (fixed per thread)
  int ra = tid >> 2, ca = (tid & 3) * 8;            // A: row ra, cols ca..ca+7
  int kb2 = (tid >> 4) * 2, ng = (tid & 15) * 4;    // B: k rows kb2,kb2+1, cols ng..ng+3
  const float* pA  = A + (size_t)(m0 + ra) * lda + ca;
  const float* pB0 = Bw + (size_t)kb2 * ldb + n0 + ng;
  const float* pB1 = pB0 + ldb;
  bf16* sA = &As[ra * 32 + ca];

  f32x8 acc0 = {}, acc1 = {};
  for (int k0 = 0; k0 < K; k0 += 32) {
    float4 a0 = *(const float4*)pA;
    float4 a1 = *(const float4*)(pA + 4);
    float4 b0 = *(const float4*)pB0;
    float4 b1 = *(const float4*)pB1;
    pA += 32; pB0 += (size_t)32 * ldb; pB1 += (size_t)32 * ldb;
    __builtin_prefetch(pB0, 0, 1);
    __syncthreads();                       // prev tile consumed
    *(bf16x8*)sA = pack8(a0, a1);
    bf16x2 w0 = {(bf16)b0.x, (bf16)b1.x}; *(bf16x2*)&Bt[(ng + 0) * 32 + kb2] = w0;
    bf16x2 w1 = {(bf16)b0.y, (bf16)b1.y}; *(bf16x2*)&Bt[(ng + 1) * 32 + kb2] = w1;
    bf16x2 w2 = {(bf16)b0.z, (bf16)b1.z}; *(bf16x2*)&Bt[(ng + 2) * 32 + kb2] = w2;
    bf16x2 w3 = {(bf16)b0.w, (bf16)b1.w}; *(bf16x2*)&Bt[(ng + 3) * 32 + kb2] = w3;
    __syncthreads();
    bf16x16 a = load_a_frag(As, 32, wrow * 16, 0, lane);
    acc0 = wmma_bf16(a, load_b_frag(Bt, 32, wcol * 32 + 0,  0, lane), acc0);
    acc1 = wmma_bf16(a, load_b_frag(Bt, 32, wcol * 32 + 16, 0, lane), acc1);
  }
#pragma unroll
  for (int r = 0; r < 8; ++r) {
    int m = m0 + wrow * 16 + r + ((lane & 16) ? 8 : 0);
    int nA = n0 + wcol * 32 + (lane & 15);
    int nB = nA + 16;
    float vA = acc0[r] + bias[nA];
    float vB = acc1[r] + bias[nB];
    if (resid) { vA += resid[(size_t)m * N + nA]; vB += resid[(size_t)m * N + nB]; }
    C0[(size_t)m * N + nA] = vA;
    C0[(size_t)m * N + nB] = vB;
    if (C1) { C1[(size_t)m * N + nA] = vA; C1[(size_t)m * N + nB] = vB; }
  }
}

// ---------------- flash attention (GQA), 64 queries/block, 4 waves x 16 rows ----------------
__global__ __launch_bounds__(128) void attn_k(const float* __restrict__ qb,
                                              const float* __restrict__ kb,
                                              const float* __restrict__ vb,
                                              float* __restrict__ ob) {
  __shared__ __align__(32) bf16 Qs[64 * 64];        // [q][d]
  __shared__ __align__(32) bf16 Ks[64 * 64];        // [key][d]  ([n][k] for scores)
  __shared__ __align__(32) bf16 Vt[64 * 64];        // [d][key]  ([n][k] for PV)
  __shared__ __align__(32) bf16 Ps[4 * 16 * 64];    // per-wave P tile [m][key]

  int tid = threadIdx.x, lane = tid & 31, wid = tid >> 5;
  int q0 = blockIdx.x * 64;
  int bh = blockIdx.y;
  int bb = bh >> 4, hq = bh & 15, kvh = hq >> 2;
  const float* qp = qb + (size_t)(bb * Sc) * (HQc * Dc) + hq * Dc;
  const float* kp = kb + (size_t)(bb * Sc) * (HKVc * Dc) + kvh * Dc;
  const float* vp = vb + (size_t)(bb * Sc) * (HKVc * Dc) + kvh * Dc;

  // staging maps
  int rq = tid >> 1, cq = (tid & 1) * 32;            // rows of Q/K tiles
  int kp2 = (tid >> 2) * 2, db = (tid & 3) * 16;     // V transpose: key pair, d group
  {
    const float* p = qp + (size_t)(q0 + rq) * (HQc * Dc) + cq;
#pragma unroll
    for (int cc = 0; cc < 32; cc += 8)
      *(bf16x8*)&Qs[rq * 64 + cq + cc] =
          pack8(*(const float4*)(p + cc), *(const float4*)(p + cc + 4));
  }
  __syncthreads();
  bf16x16 aq0 = load_a_frag(Qs, 64, wid * 16, 0, lane);
  bf16x16 aq1 = load_a_frag(Qs, 64, wid * 16, 32, lane);

  const float* pk  = kp + (size_t)rq * (HKVc * Dc) + cq;
  const float* pv0 = vp + (size_t)kp2 * (HKVc * Dc) + db;
  const float* pv1 = pv0 + (HKVc * Dc);

  f32x8 o0 = {}, o1 = {}, o2 = {}, o3 = {};
  float runmax[8], runsum[8];
#pragma unroll
  for (int r = 0; r < 8; ++r) { runmax[r] = -3.0e38f; runsum[r] = 0.f; }
  const float sc = 0.125f;   // 1/sqrt(64)

  for (int kt = 0; kt < Sc; kt += 64) {
    __syncthreads();
    // K tile: [key][d] row-major
#pragma unroll
    for (int cc = 0; cc < 32; cc += 8)
      *(bf16x8*)&Ks[rq * 64 + cq + cc] =
          pack8(*(const float4*)(pk + cc), *(const float4*)(pk + cc + 4));
    // V tile transposed: Vt[d][key], two keys packed per 32-bit store
#pragma unroll
    for (int dd = 0; dd < 16; dd += 4) {
      float4 va = *(const float4*)(pv0 + dd);
      float4 vc = *(const float4*)(pv1 + dd);
      bf16x2 t0 = {(bf16)va.x, (bf16)vc.x}; *(bf16x2*)&Vt[(db + dd + 0) * 64 + kp2] = t0;
      bf16x2 t1 = {(bf16)va.y, (bf16)vc.y}; *(bf16x2*)&Vt[(db + dd + 1) * 64 + kp2] = t1;
      bf16x2 t2 = {(bf16)va.z, (bf16)vc.z}; *(bf16x2*)&Vt[(db + dd + 2) * 64 + kp2] = t2;
      bf16x2 t3 = {(bf16)va.w, (bf16)vc.w}; *(bf16x2*)&Vt[(db + dd + 3) * 64 + kp2] = t3;
    }
    pk += (size_t)64 * (HKVc * Dc);
    pv0 += (size_t)64 * (HKVc * Dc);
    pv1 += (size_t)64 * (HKVc * Dc);
    __syncthreads();

    f32x8 s0 = {}, s1 = {}, s2 = {}, s3 = {};
    s0 = wmma_bf16(aq0, load_b_frag(Ks, 64,  0, 0, lane), s0);
    s0 = wmma_bf16(aq1, load_b_frag(Ks, 64,  0, 32, lane), s0);
    s1 = wmma_bf16(aq0, load_b_frag(Ks, 64, 16, 0, lane), s1);
    s1 = wmma_bf16(aq1, load_b_frag(Ks, 64, 16, 32, lane), s1);
    s2 = wmma_bf16(aq0, load_b_frag(Ks, 64, 32, 0, lane), s2);
    s2 = wmma_bf16(aq1, load_b_frag(Ks, 64, 32, 32, lane), s2);
    s3 = wmma_bf16(aq0, load_b_frag(Ks, 64, 48, 0, lane), s3);
    s3 = wmma_bf16(aq1, load_b_frag(Ks, 64, 48, 32, lane), s3);

    // online softmax per row (row r <-> acc element r, across the 16-lane half)
#pragma unroll
    for (int r = 0; r < 8; ++r) {
      float tm = fmaxf(fmaxf(s0[r], s1[r]), fmaxf(s2[r], s3[r]));
      tm = half16_max(tm);
      float nm = fmaxf(runmax[r], tm);
      float corr = __expf((runmax[r] - nm) * sc);
      runmax[r] = nm;
      runsum[r] *= corr;
      o0[r] *= corr; o1[r] *= corr; o2[r] *= corr; o3[r] *= corr;
      float p0 = __expf((s0[r] - nm) * sc);
      float p1 = __expf((s1[r] - nm) * sc);
      float p2 = __expf((s2[r] - nm) * sc);
      float p3 = __expf((s3[r] - nm) * sc);
      runsum[r] += half16_sum(p0 + p1 + p2 + p3);
      int m = r + ((lane & 16) ? 8 : 0);
      int col = lane & 15;
      bf16* pw = &Ps[wid * 16 * 64 + m * 64];
      pw[ 0 + col] = (bf16)p0;
      pw[16 + col] = (bf16)p1;
      pw[32 + col] = (bf16)p2;
      pw[48 + col] = (bf16)p3;
    }

    // O += P @ V
    bf16x16 ap0 = load_a_frag(&Ps[wid * 16 * 64], 64, 0, 0, lane);
    bf16x16 ap1 = load_a_frag(&Ps[wid * 16 * 64], 64, 0, 32, lane);
    o0 = wmma_bf16(ap0, load_b_frag(Vt, 64,  0, 0, lane), o0);
    o0 = wmma_bf16(ap1, load_b_frag(Vt, 64,  0, 32, lane), o0);
    o1 = wmma_bf16(ap0, load_b_frag(Vt, 64, 16, 0, lane), o1);
    o1 = wmma_bf16(ap1, load_b_frag(Vt, 64, 16, 32, lane), o1);
    o2 = wmma_bf16(ap0, load_b_frag(Vt, 64, 32, 0, lane), o2);
    o2 = wmma_bf16(ap1, load_b_frag(Vt, 64, 32, 32, lane), o2);
    o3 = wmma_bf16(ap0, load_b_frag(Vt, 64, 48, 0, lane), o3);
    o3 = wmma_bf16(ap1, load_b_frag(Vt, 64, 48, 32, lane), o3);
  }

#pragma unroll
  for (int r = 0; r < 8; ++r) {
    float inv = 1.f / runsum[r];
    int m = wid * 16 + r + ((lane & 16) ? 8 : 0);
    int col = lane & 15;
    size_t ro = (size_t)(bb * Sc + q0 + m) * (HQc * Dc) + hq * Dc;
    ob[ro +  0 + col] = o0[r] * inv;
    ob[ro + 16 + col] = o1[r] * inv;
    ob[ro + 32 + col] = o2[r] * inv;
    ob[ro + 48 + col] = o3[r] * inv;
  }
}

// ---------------- router: logits, top-2, softmax, counts ----------------
__global__ __launch_bounds__(64) void router_k(const float* __restrict__ t,
                                               const float* __restrict__ rw,
                                               const float* __restrict__ rb,
                                               int* __restrict__ counts,
                                               int* __restrict__ tokexp,
                                               float* __restrict__ tokw) {
  int tok = blockIdx.x, tid = threadIdx.x, lane = tid & 31, wid = tid >> 5;
  const float* xr = t + (size_t)tok * Hc;
  float acc[Ec];
#pragma unroll
  for (int e = 0; e < Ec; ++e) acc[e] = 0.f;
  for (int h = tid; h < Hc; h += 64) {
    float xv = xr[h];
    float4 w0 = *(const float4*)(rw + h * Ec);
    float4 w1 = *(const float4*)(rw + h * Ec + 4);
    acc[0] += xv * w0.x; acc[1] += xv * w0.y; acc[2] += xv * w0.z; acc[3] += xv * w0.w;
    acc[4] += xv * w1.x; acc[5] += xv * w1.y; acc[6] += xv * w1.z; acc[7] += xv * w1.w;
  }
#pragma unroll
  for (int e = 0; e < Ec; ++e)
#pragma unroll
    for (int m = 16; m >= 1; m >>= 1) acc[e] += __shfl_xor(acc[e], m, 32);
  __shared__ float part[2][Ec];
  if (lane == 0)
#pragma unroll
    for (int e = 0; e < Ec; ++e) part[wid][e] = acc[e];
  __syncthreads();
  if (tid == 0) {
    float lg[Ec];
#pragma unroll
    for (int e = 0; e < Ec; ++e) lg[e] = part[0][e] + part[1][e] + rb[e];
    int i0 = 0;
    for (int e = 1; e < Ec; ++e) if (lg[e] > lg[i0]) i0 = e;
    int i1 = -1;
    for (int e = 0; e < Ec; ++e) { if (e == i0) continue; if (i1 < 0 || lg[e] > lg[i1]) i1 = e; }
    float mx = fmaxf(lg[i0], lg[i1]);
    float e0 = __expf(lg[i0] - mx), e1 = __expf(lg[i1] - mx);
    float den = e0 + e1;
    tokexp[tok * 2 + 0] = i0; tokw[tok * 2 + 0] = e0 / den;
    tokexp[tok * 2 + 1] = i1; tokw[tok * 2 + 1] = e1 / den;
    atomicAdd(&counts[i0], 1);
    atomicAdd(&counts[i1], 1);
  }
}

__global__ void init_k(int* counts, int* cursors, int* gtok, float* gwt) {
  int i = blockIdx.x * 256 + threadIdx.x;
  if (i < Ec) { counts[i] = 0; cursors[i] = 0; }
  if (i < MAXROWS) { gtok[i] = -1; gwt[i] = 0.f; }
}

__global__ void plan_k(const int* counts, int* tile_e, int* tile_rb, int* offs) {
  if (blockIdx.x == 0 && threadIdx.x == 0) {
    int base = 0, tile = 0;
    for (int e = 0; e < Ec; ++e) {
      offs[e] = base;
      int nt = (counts[e] + 63) >> 6;
      for (int t2 = 0; t2 < nt; ++t2) { tile_e[tile] = e; tile_rb[tile] = base + t2 * 64; ++tile; }
      base += nt * 64;
    }
    for (; tile < MAXTILES; ++tile) { tile_e[tile] = -1; tile_rb[tile] = 0; }
  }
}

__global__ __launch_bounds__(256) void scatter_k(const int* tokexp, const float* tokw,
                                                 const int* offs, int* cursors,
                                                 int* gtok, float* gwt) {
  int tok = blockIdx.x * 256 + threadIdx.x;
  if (tok >= Tc) return;
#pragma unroll
  for (int k2 = 0; k2 < TOPKc; ++k2) {
    int e = tokexp[tok * 2 + k2];
    int pos = atomicAdd(&cursors[e], 1);
    int row = offs[e] + pos;
    gtok[row] = tok;
    gwt[row] = tokw[tok * 2 + k2];
  }
}

// ---------------- MoE grouped gate/up (fused) ----------------
__global__ __launch_bounds__(256) void moe_gateup_k(const float* __restrict__ t,
                                                    const float* __restrict__ gate_w,
                                                    const float* __restrict__ gate_b,
                                                    const float* __restrict__ up_w,
                                                    const float* __restrict__ up_b,
                                                    const int* __restrict__ tile_e,
                                                    const int* __restrict__ tile_rb,
                                                    const int* __restrict__ gtok,
                                                    bf16* __restrict__ act) {
  int e = tile_e[blockIdx.y];
  if (e < 0) return;
  int rbase = tile_rb[blockIdx.y];
  int n0 = blockIdx.x * 64;
  __shared__ __align__(32) bf16 As[64 * 32];
  __shared__ __align__(32) bf16 Gt[64 * 32];
  __shared__ __align__(32) bf16 Ut[64 * 32];
  int tid = threadIdx.x, lane = tid & 31, wid = tid >> 5;
  int wrow = wid >> 1, wcol = wid & 1;

  int ra = tid >> 2, ca = (tid & 3) * 8;
  int kb2 = (tid >> 4) * 2, ng = (tid & 15) * 4;
  int tk = gtok[rbase + ra];
  const float* pA = (tk >= 0) ? (t + (size_t)tk * Hc + ca) : t;  // dummy if invalid
  const float* gw = gate_w + (size_t)e * Hc * Ic;
  const float* uw = up_w + (size_t)e * Hc * Ic;
  const float* pG0 = gw + (size_t)kb2 * Ic + n0 + ng;
  const float* pG1 = pG0 + Ic;
  const float* pU0 = uw + (size_t)kb2 * Ic + n0 + ng;
  const float* pU1 = pU0 + Ic;
  bf16* sA = &As[ra * 32 + ca];

  f32x8 g0 = {}, g1 = {}, u0 = {}, u1 = {};
  for (int k0 = 0; k0 < Hc; k0 += 32) {
    float4 z = {0.f, 0.f, 0.f, 0.f};
    float4 a0 = (tk >= 0) ? *(const float4*)pA : z;
    float4 a1 = (tk >= 0) ? *(const float4*)(pA + 4) : z;
    float4 gg0 = *(const float4*)pG0;
    float4 gg1 = *(const float4*)pG1;
    float4 uu0 = *(const float4*)pU0;
    float4 uu1 = *(const float4*)pU1;
    pA += 32;
    pG0 += (size_t)32 * Ic; pG1 += (size_t)32 * Ic;
    pU0 += (size_t)32 * Ic; pU1 += (size_t)32 * Ic;
    __builtin_prefetch(pG0, 0, 1);
    __builtin_prefetch(pU0, 0, 1);
    __syncthreads();
    *(bf16x8*)sA = pack8(a0, a1);
    bf16x2 wg0 = {(bf16)gg0.x, (bf16)gg1.x}; *(bf16x2*)&Gt[(ng + 0) * 32 + kb2] = wg0;
    bf16x2 wg1 = {(bf16)gg0.y, (bf16)gg1.y}; *(bf16x2*)&Gt[(ng + 1) * 32 + kb2] = wg1;
    bf16x2 wg2 = {(bf16)gg0.z, (bf16)gg1.z}; *(bf16x2*)&Gt[(ng + 2) * 32 + kb2] = wg2;
    bf16x2 wg3 = {(bf16)gg0.w, (bf16)gg1.w}; *(bf16x2*)&Gt[(ng + 3) * 32 + kb2] = wg3;
    bf16x2 wu0 = {(bf16)uu0.x, (bf16)uu1.x}; *(bf16x2*)&Ut[(ng + 0) * 32 + kb2] = wu0;
    bf16x2 wu1 = {(bf16)uu0.y, (bf16)uu1.y}; *(bf16x2*)&Ut[(ng + 1) * 32 + kb2] = wu1;
    bf16x2 wu2 = {(bf16)uu0.z, (bf16)uu1.z}; *(bf16x2*)&Ut[(ng + 2) * 32 + kb2] = wu2;
    bf16x2 wu3 = {(bf16)uu0.w, (bf16)uu1.w}; *(bf16x2*)&Ut[(ng + 3) * 32 + kb2] = wu3;
    __syncthreads();
    bf16x16 a = load_a_frag(As, 32, wrow * 16, 0, lane);
    g0 = wmma_bf16(a, load_b_frag(Gt, 32, wcol * 32 + 0,  0, lane), g0);
    g1 = wmma_bf16(a, load_b_frag(Gt, 32, wcol * 32 + 16, 0, lane), g1);
    u0 = wmma_bf16(a, load_b_frag(Ut, 32, wcol * 32 + 0,  0, lane), u0);
    u1 = wmma_bf16(a, load_b_frag(Ut, 32, wcol * 32 + 16, 0, lane), u1);
  }
  const float* gb = gate_b + (size_t)e * Ic;
  const float* ub = up_b + (size_t)e * Ic;
#pragma unroll
  for (int r = 0; r < 8; ++r) {
    int rloc = wrow * 16 + r + ((lane & 16) ? 8 : 0);
    int row = rbase + rloc;
    int nA = n0 + wcol * 32 + (lane & 15);
    int nB = nA + 16;
    float gA = g0[r] + gb[nA], uA = u0[r] + ub[nA];
    float gB = g1[r] + gb[nB], uB = u1[r] + ub[nB];
    float sA2 = gA / (1.f + __expf(-gA));
    float sB2 = gB / (1.f + __expf(-gB));
    act[(size_t)row * Ic + nA] = (bf16)(sA2 * uA);
    act[(size_t)row * Ic + nB] = (bf16)(sB2 * uB);
  }
}

// ---------------- MoE grouped down-proj + weighted scatter-add ----------------
__global__ __launch_bounds__(256) void moe_down_k(const bf16* __restrict__ act,
                                                  const float* __restrict__ down_w,
                                                  const float* __restrict__ down_b,
                                                  const int* __restrict__ tile_e,
                                                  const int* __restrict__ tile_rb,
                                                  const int* __restrict__ gtok,
                                                  const float* __restrict__ gwt,
                                                  float* __restrict__ out) {
  int e = tile_e[blockIdx.y];
  if (e < 0) return;
  int rbase = tile_rb[blockIdx.y];
  int n0 = blockIdx.x * 64;
  __shared__ __align__(32) bf16 As[64 * 32];
  __shared__ __align__(32) bf16 Bt[64 * 32];
  int tid = threadIdx.x, lane = tid & 31, wid = tid >> 5;
  int wrow = wid >> 1, wcol = wid & 1;

  int ra = tid >> 2, ca = (tid & 3) * 8;
  int kb2 = (tid >> 4) * 2, ng = (tid & 15) * 4;
  const bf16* pAct = act + (size_t)(rbase + ra) * Ic + ca;   // 16B per thread per step
  unsigned sAoff = (unsigned)(size_t)&As[ra * 32 + ca];      // LDS byte offset
  const float* dw = down_w + (size_t)e * Ic * Hc;
  const float* pB0 = dw + (size_t)kb2 * Hc + n0 + ng;
  const float* pB1 = pB0 + Hc;

  f32x8 acc0 = {}, acc1 = {};
  for (int k0 = 0; k0 < Ic; k0 += 32) {
    __syncthreads();                        // prev tile consumed
    // async DMA: 16 bytes of bf16 activations straight into LDS (no VGPR round-trip)
    ASYNC_CP_B128(sAoff, pAct);
    pAct += 32;
    float4 b0 = *(const float4*)pB0;
    float4 b1 = *(const float4*)pB1;
    pB0 += (size_t)32 * Hc; pB1 += (size_t)32 * Hc;
    __builtin_prefetch(pB0, 0, 1);
    bf16x2 w0 = {(bf16)b0.x, (bf16)b1.x}; *(bf16x2*)&Bt[(ng + 0) * 32 + kb2] = w0;
    bf16x2 w1 = {(bf16)b0.y, (bf16)b1.y}; *(bf16x2*)&Bt[(ng + 1) * 32 + kb2] = w1;
    bf16x2 w2 = {(bf16)b0.z, (bf16)b1.z}; *(bf16x2*)&Bt[(ng + 2) * 32 + kb2] = w2;
    bf16x2 w3 = {(bf16)b0.w, (bf16)b1.w}; *(bf16x2*)&Bt[(ng + 3) * 32 + kb2] = w3;
    WAIT_ASYNC0();
    __syncthreads();
    bf16x16 a = load_a_frag(As, 32, wrow * 16, 0, lane);
    acc0 = wmma_bf16(a, load_b_frag(Bt, 32, wcol * 32 + 0,  0, lane), acc0);
    acc1 = wmma_bf16(a, load_b_frag(Bt, 32, wcol * 32 + 16, 0, lane), acc1);
  }
  const float* db = down_b + (size_t)e * Hc;
#pragma unroll
  for (int r = 0; r < 8; ++r) {
    int rloc = wrow * 16 + r + ((lane & 16) ? 8 : 0);
    int row = rbase + rloc;
    int tk = gtok[row];
    if (tk < 0) continue;
    float w = gwt[row];
    int nA = n0 + wcol * 32 + (lane & 15);
    int nB = nA + 16;
    atomicAdd(&out[(size_t)tk * Hc + nA], w * (acc0[r] + db[nA]));
    atomicAdd(&out[(size_t)tk * Hc + nB], w * (acc1[r] + db[nB]));
  }
}

// ---------------- launcher ----------------
extern "C" void kernel_launch(void* const* d_in, const int* in_sizes, int n_in,
                              void* d_out, int out_size, void* d_ws, size_t ws_size,
                              hipStream_t stream) {
  const float* x        = (const float*)d_in[0];
  const float* norm1_w  = (const float*)d_in[1];
  const float* q_w      = (const float*)d_in[2];
  const float* q_b      = (const float*)d_in[3];
  const float* k_w      = (const float*)d_in[4];
  const float* k_b      = (const float*)d_in[5];
  const float* v_w      = (const float*)d_in[6];
  const float* v_b      = (const float*)d_in[7];
  const float* o_w      = (const float*)d_in[8];
  const float* o_b      = (const float*)d_in[9];
  const float* norm2_w  = (const float*)d_in[10];
  const float* router_w = (const float*)d_in[11];
  const float* router_b = (const float*)d_in[12];
  const float* gate_w   = (const float*)d_in[13];
  const float* gate_b   = (const float*)d_in[14];
  const float* up_w     = (const float*)d_in[15];
  const float* up_b     = (const float*)d_in[16];
  const float* down_w   = (const float*)d_in[17];
  const float* down_b   = (const float*)d_in[18];
  float* dout = (float*)d_out;

  char* w = (char*)d_ws;
  size_t off = 0;
  auto take = [&](size_t bytes) { size_t o = off; off = (off + bytes + 255) & ~(size_t)255; return o; };
  float* h    = (float*)(w + take((size_t)Tc * Hc * 4));
  float* qb   = (float*)(w + take((size_t)Tc * HQc * Dc * 4));
  float* kb   = (float*)(w + take((size_t)Tc * HKVc * Dc * 4));
  float* vb   = (float*)(w + take((size_t)Tc * HKVc * Dc * 4));
  float* oat  = (float*)(w + take((size_t)Tc * HQc * Dc * 4));
  float* x1   = (float*)(w + take((size_t)Tc * Hc * 4));
  float* tn   = (float*)(w + take((size_t)Tc * Hc * 4));
  bf16*  act  = (bf16*) (w + take((size_t)MAXROWS * Ic * 2));
  int*   gtok = (int*)  (w + take((size_t)MAXROWS * 4));
  float* gwt  = (float*)(w + take((size_t)MAXROWS * 4));
  int*   tokexp = (int*)  (w + take((size_t)Tc * 2 * 4));
  float* tokw   = (float*)(w + take((size_t)Tc * 2 * 4));
  int*   counts = (int*)(w + take(Ec * 4));
  int*   cursors= (int*)(w + take(Ec * 4));
  int*   offs   = (int*)(w + take(16 * 4));
  int*   tile_e = (int*)(w + take(MAXTILES * 4));
  int*   tile_rb= (int*)(w + take(MAXTILES * 4));

  init_k<<<dim3((MAXROWS + 255) / 256), 256, 0, stream>>>(counts, cursors, gtok, gwt);
  rmsnorm_k<<<Tc, 256, 0, stream>>>(x, norm1_w, h);

  gemm_bf16_k<<<dim3((HQc * Dc) / 64, Tc / 64), 256, 0, stream>>>(
      h, Hc, q_w, HQc * Dc, q_b, nullptr, qb, nullptr, HQc * Dc, Hc);
  gemm_bf16_k<<<dim3((HKVc * Dc) / 64, Tc / 64), 256, 0, stream>>>(
      h, Hc, k_w, HKVc * Dc, k_b, nullptr, kb, nullptr, HKVc * Dc, Hc);
  gemm_bf16_k<<<dim3((HKVc * Dc) / 64, Tc / 64), 256, 0, stream>>>(
      h, Hc, v_w, HKVc * Dc, v_b, nullptr, vb, nullptr, HKVc * Dc, Hc);

  attn_k<<<dim3(Sc / 64, Bc * HQc), 128, 0, stream>>>(qb, kb, vb, oat);

  // o-proj + residual; writes x1 and initializes d_out = x1
  gemm_bf16_k<<<dim3(Hc / 64, Tc / 64), 256, 0, stream>>>(
      oat, HQc * Dc, o_w, Hc, o_b, x, x1, dout, Hc, HQc * Dc);

  rmsnorm_k<<<Tc, 256, 0, stream>>>(x1, norm2_w, tn);
  router_k<<<Tc, 64, 0, stream>>>(tn, router_w, router_b, counts, tokexp, tokw);
  plan_k<<<1, 1, 0, stream>>>(counts, tile_e, tile_rb, offs);
  scatter_k<<<dim3(Tc / 256), 256, 0, stream>>>(tokexp, tokw, offs, cursors, gtok, gwt);

  moe_gateup_k<<<dim3(Ic / 64, MAXTILES), 256, 0, stream>>>(
      tn, gate_w, gate_b, up_w, up_b, tile_e, tile_rb, gtok, act);
  moe_down_k<<<dim3(Hc / 64, MAXTILES), 256, 0, stream>>>(
      act, down_w, down_b, tile_e, tile_rb, gtok, gwt, dout);
}
